// CPlayer_90082644066621
// MI455X (gfx1250) — matrix operand order repaired and checked by reading, not compiled, not run on hardware
//
#include <hip/hip_runtime.h>

#define NNODES 100000
#define NEDGES 1600000
#define INFEA  128
#define HIDDEN 128
#define RANK   64
#define RTILES (NNODES / 16)    // 6250 (exact)
#define RJOBS  (RTILES / 2)     // 3125 double-strip jobs (exact)

typedef __attribute__((ext_vector_type(16))) __bf16 v16bf;
typedef __attribute__((ext_vector_type(8)))  float  v8f;

__device__ __forceinline__ unsigned short f2bf_bits(float f) {
    union { float f; unsigned int u; } c; c.f = f;
    unsigned int u = c.u;
    u += 0x7FFFu + ((u >> 16) & 1u);           // RNE (host-side style, used in pack only)
    return (unsigned short)(u >> 16);
}

// ---------------------------------------------------------------------------
// Pack W and V into the WMMA B-operand per-lane layout (bf16).
// B block (32x16): lane&15 = column, element e -> K = (lane>>4)*8 + (e&7) + (e>>3)*16.
// wp: [nt(4)][kb(4)][lane(32)][e(16)]   B[k][n] = W[k][n]         (h = x @ W)
// vp: [nt(8)][kb(2)][lane(32)][e(16)]   B[k][n] = V[n][k]         (out = t @ V^T)
// ---------------------------------------------------------------------------
__global__ void pack_weights(const float* __restrict__ W, const float* __restrict__ V,
                             unsigned short* __restrict__ wp, unsigned short* __restrict__ vp) {
    int i = blockIdx.x * blockDim.x + threadIdx.x;
    if (i >= 8192) return;
    int e    = i & 15;
    int lane = (i >> 4) & 31;
    int col  = lane & 15;
    int klocal = ((lane >> 4) << 3) + (e & 7) + ((e >> 3) << 4);       // 0..31 within K-block
    int blk = i >> 9;                          // 0..15
    {   // W pack: blk = nt*4 + kb
        int kb = blk & 3, nt = blk >> 2;
        wp[i] = f2bf_bits(W[(kb * 32 + klocal) * RANK + nt * 16 + col]);
    }
    {   // V pack: blk = nt*2 + kb
        int kb = blk & 1, nt = blk >> 1;
        vp[i] = f2bf_bits(V[(nt * 16 + col) * RANK + kb * 32 + klocal]);
    }
}

// Build an A-operand vector (16-bit A 16x32 layout) from 32 f32, native cvt.
__device__ __forceinline__ v16bf load_a(const float* __restrict__ p) {
    v16bf a;
    #pragma unroll
    for (int e = 0; e < 8; ++e) a[e]     = (__bf16)p[e];
    #pragma unroll
    for (int e = 0; e < 8; ++e) a[8 + e] = (__bf16)p[16 + e];
    return a;
}

// ---------------------------------------------------------------------------
// GEMM1: h[N,64] = x[N,128] @ W[128,64]   (bf16 WMMA, f32 accum)
// One wave per 32-row job (2 strips); each B block feeds 2 WMMAs.
// ---------------------------------------------------------------------------
__global__ void gemm_h(const float* __restrict__ x, const unsigned short* __restrict__ wp,
                       float* __restrict__ h) {
    int lane = threadIdx.x & 31;
    int job  = blockIdx.x * (blockDim.x >> 5) + (threadIdx.x >> 5);
    if (job >= RJOBS) return;
    int row0  = job * 32;
    int arow  = row0 + (lane & 15);
    int khalf = (lane >> 4) * 8;

    v8f c0[4], c1[4];
    #pragma unroll
    for (int nt = 0; nt < 4; ++nt) { c0[nt] = (v8f)(0.0f); c1[nt] = (v8f)(0.0f); }

    #pragma unroll
    for (int kb = 0; kb < 4; ++kb) {
        v16bf a0 = load_a(x + (size_t)arow        * INFEA + kb * 32 + khalf);
        v16bf a1 = load_a(x + (size_t)(arow + 16) * INFEA + kb * 32 + khalf);
        #pragma unroll
        for (int nt = 0; nt < 4; ++nt) {
            v16bf b = *(const v16bf*)(wp + (((nt * 4 + kb) * 32 + lane) << 4));
            c0[nt] = __builtin_amdgcn_wmma_f32_16x16x32_bf16(
                         false, a0, false, b, (short)0, c0[nt], false, false);
            c1[nt] = __builtin_amdgcn_wmma_f32_16x16x32_bf16(
                         false, a1, false, b, (short)0, c1[nt], false, false);
        }
    }
    int coladd = lane & 15;
    int rowadd = (lane >> 4) * 8;
    #pragma unroll
    for (int nt = 0; nt < 4; ++nt)
        #pragma unroll
        for (int j = 0; j < 8; ++j) {
            h[(size_t)(row0      + rowadd + j) * RANK + nt * 16 + coladd] = c0[nt][j];
            h[(size_t)(row0 + 16 + rowadd + j) * RANK + nt * 16 + coladd] = c1[nt][j];
        }
}

// ---------------------------------------------------------------------------
// Zero accumulators (logp then negcnt, contiguous 2*N*64 f32) as float4.
// ---------------------------------------------------------------------------
__global__ void zero_acc(float4* __restrict__ p, int n4) {
    int i = blockIdx.x * blockDim.x + threadIdx.x;
    if (i < n4) p[i] = make_float4(0.f, 0.f, 0.f, 0.f);
}

// ---------------------------------------------------------------------------
// Edge scatter: thread = (edge, rank-pair). Reads float2 of gathered h row,
// L2-resident global_atomic_add_f32 into log/neg accumulators.
// ---------------------------------------------------------------------------
__global__ void edge_scatter(const float* __restrict__ h,
                             const int* __restrict__ src, const int* __restrict__ dst,
                             float* __restrict__ logp, float* __restrict__ ncnt) {
    long long t = (long long)blockIdx.x * blockDim.x + threadIdx.x;
    int rp = (int)(t & 31);                    // rank pair 0..31
    long long e = t >> 5;
    if (e >= NEDGES) return;
    int s = src[e], d = dst[e];
    float2 v = *(const float2*)(h + (size_t)s * RANK + rp * 2);
    size_t o = (size_t)d * RANK + rp * 2;
    atomicAdd(&logp[o],     __logf(fabsf(v.x)));
    atomicAdd(&logp[o + 1], __logf(fabsf(v.y)));
    if (v.x < 0.0f) atomicAdd(&ncnt[o],     1.0f);
    if (v.y < 0.0f) atomicAdd(&ncnt[o + 1], 1.0f);
}

// ---------------------------------------------------------------------------
// Finalize: trans = norm * sign * exp(logp), in place over logp.
// ---------------------------------------------------------------------------
__global__ void finalize(const float* __restrict__ norm,
                         float* __restrict__ logp, const float* __restrict__ ncnt) {
    int i = blockIdx.x * blockDim.x + threadIdx.x;
    if (i >= NNODES * RANK) return;
    int n = i >> 6;
    float sign = 1.0f - 2.0f * fmodf(ncnt[i], 2.0f);
    logp[i] = norm[n] * sign * __expf(logp[i]);
}

// ---------------------------------------------------------------------------
// GEMM2: out[N,128] = trans[N,64] @ V^T   (bf16 WMMA, f32 accum)
// One wave per 32-row job; 8 N-tiles, 2 K-blocks; each B feeds 2 WMMAs.
// ---------------------------------------------------------------------------
__global__ void gemm_out(const float* __restrict__ trans, const unsigned short* __restrict__ vp,
                         float* __restrict__ out) {
    int lane = threadIdx.x & 31;
    int job  = blockIdx.x * (blockDim.x >> 5) + (threadIdx.x >> 5);
    if (job >= RJOBS) return;
    int row0  = job * 32;
    int arow  = row0 + (lane & 15);
    int khalf = (lane >> 4) * 8;

    v8f c0[8], c1[8];
    #pragma unroll
    for (int nt = 0; nt < 8; ++nt) { c0[nt] = (v8f)(0.0f); c1[nt] = (v8f)(0.0f); }

    #pragma unroll
    for (int kb = 0; kb < 2; ++kb) {
        v16bf a0 = load_a(trans + (size_t)arow        * RANK + kb * 32 + khalf);
        v16bf a1 = load_a(trans + (size_t)(arow + 16) * RANK + kb * 32 + khalf);
        #pragma unroll
        for (int nt = 0; nt < 8; ++nt) {
            v16bf b = *(const v16bf*)(vp + (((nt * 2 + kb) * 32 + lane) << 4));
            c0[nt] = __builtin_amdgcn_wmma_f32_16x16x32_bf16(
                         false, a0, false, b, (short)0, c0[nt], false, false);
            c1[nt] = __builtin_amdgcn_wmma_f32_16x16x32_bf16(
                         false, a1, false, b, (short)0, c1[nt], false, false);
        }
    }
    int coladd = lane & 15;
    int rowadd = (lane >> 4) * 8;
    #pragma unroll
    for (int nt = 0; nt < 8; ++nt)
        #pragma unroll
        for (int j = 0; j < 8; ++j) {
            out[(size_t)(row0      + rowadd + j) * HIDDEN + nt * 16 + coladd] = c0[nt][j];
            out[(size_t)(row0 + 16 + rowadd + j) * HIDDEN + nt * 16 + coladd] = c1[nt][j];
        }
}

extern "C" void kernel_launch(void* const* d_in, const int* in_sizes, int n_in,
                              void* d_out, int out_size, void* d_ws, size_t ws_size,
                              hipStream_t stream) {
    const float* x    = (const float*)d_in[0];
    const float* norm = (const float*)d_in[1];
    const float* W    = (const float*)d_in[2];
    const float* V    = (const float*)d_in[3];
    const int*   src  = (const int*)d_in[4];
    const int*   dst  = (const int*)d_in[5];
    float* out = (float*)d_out;

    char* ws = (char*)d_ws;
    const size_t NR_BYTES = (size_t)NNODES * RANK * sizeof(float);   // 25.6 MB
    float* h    = (float*)(ws);
    float* logp = (float*)(ws + NR_BYTES);          // later holds trans_x
    float* ncnt = (float*)(ws + 2 * NR_BYTES);
    unsigned short* wp = (unsigned short*)(ws + 3 * NR_BYTES);
    unsigned short* vp = (unsigned short*)(ws + 3 * NR_BYTES + 8192 * sizeof(unsigned short));

    pack_weights<<<32, 256, 0, stream>>>(W, V, wp, vp);

    int n4 = 2 * NNODES * RANK / 4;                                  // 3.2M float4
    zero_acc<<<(n4 + 255) / 256, 256, 0, stream>>>((float4*)logp, n4);

    gemm_h<<<(RJOBS + 7) / 8, 256, 0, stream>>>(x, wp, h);

    {
        long long total = (long long)NEDGES * 32;                    // (edge, rank-pair)
        int blocks = (int)((total + 255) / 256);
        edge_scatter<<<blocks, 256, 0, stream>>>(h, src, dst, logp, ncnt);
    }

    finalize<<<(NNODES * RANK + 255) / 256, 256, 0, stream>>>(norm, logp, ncnt);

    gemm_out<<<(RJOBS + 7) / 8, 256, 0, stream>>>(logp, vp, out);
}
